// IDGCN_73770358276516
// MI455X (gfx1250) — matrix operand order repaired and checked by reading, not compiled, not run on hardware
//
#include <hip/hip_runtime.h>
#include <hip/hip_bf16.h>
#include <math.h>

// ---------------------------------------------------------------------------
// IDGCN forward for MI455X (gfx1250, wave32, WMMA + async/TDM data movement).
// GEMM work on v_wmma_f32_16x16x32_f16; LDS staging via
// global_load_async_to_lds_b128 (ASYNCcnt) and tensor_load_to_lds (TENSORcnt).
// ---------------------------------------------------------------------------

typedef _Float16 v16h __attribute__((ext_vector_type(16)));
typedef _Float16 v8h  __attribute__((ext_vector_type(8)));
typedef float    v8f  __attribute__((ext_vector_type(8)));
typedef unsigned int v4u __attribute__((ext_vector_type(4)));
typedef int      v4i __attribute__((ext_vector_type(4)));
typedef int      v8i __attribute__((ext_vector_type(8)));

#define DEV __device__ __forceinline__

static constexpr int BB    = 64;   // batch
static constexpr int CC    = 64;   // channels
static constexpr int NN    = 170;  // nodes
static constexpr int LH    = 24;   // half length
static constexpr int TOPK_ = 136;

DEV v8f wmma_f16(v16h a, v16h b, v8f c) {
  return __builtin_amdgcn_wmma_f32_16x16x32_f16(
      false, a, false, b, (short)0, c, false, false);
}

// A-fragment (16x32 f16) per-lane K index for half h (0..15)
DEV int klocal(int lane, int h) { return h + (lane >= 16 ? 16 : 0); }

DEV float wsumf(float v) {
#pragma unroll
  for (int o = 16; o > 0; o >>= 1) v += __shfl_xor(v, o, 32);
  return v;
}
DEV float wmaxf(float v) {
#pragma unroll
  for (int o = 16; o > 0; o >>= 1) v = fmaxf(v, __shfl_xor(v, o, 32));
  return v;
}
DEV int wsumi(int v) {
#pragma unroll
  for (int o = 16; o > 0; o >>= 1) v += __shfl_xor(v, o, 32);
  return v;
}

// ---- CDNA5 async global->LDS copy (ASYNCcnt) ------------------------------
// LDS flat address low 32 bits are the LDS byte offset (aperture truncation).
DEV void async_copy_b128(unsigned lds_off, const void* gaddr) {
  asm volatile("global_load_async_to_lds_b128 %0, %1, off"
               :: "v"(lds_off), "v"(gaddr)
               : "memory");
}
DEV void wait_asynccnt0() { asm volatile("s_wait_asynccnt 0x0" ::: "memory"); }

// ---- CDNA5 Tensor Data Mover: 1D contiguous tile of f16 into LDS ----------
DEV void tdm_load_1d_f16(unsigned lds_off, const void* gptr, unsigned nelem) {
  unsigned long long ga = (unsigned long long)gptr;
  v4u g0;
  g0.x = 1u;                                                // count=1, user mode
  g0.y = lds_off;                                           // lds_addr (bytes)
  g0.z = (unsigned)ga;                                      // global_addr[31:0]
  g0.w = (unsigned)((ga >> 32) & 0x01FFFFFFu) | (2u << 30); // addr[56:32], type=2
  v8i g1;
  g1[0] = (int)(1u << 16);                        // wg_mask=0, data_size=1 (2B)
  g1[1] = (int)((nelem & 0xFFFFu) << 16);         // tensor_dim0 lo16
  g1[2] = (int)(((nelem >> 16) & 0xFFFFu) | (1u << 16));  // dim0 hi | tensor_dim1=1 lo
  g1[3] = (int)((nelem & 0xFFFFu) << 16);         // dim1 hi=0 | tile_dim0
  g1[4] = 1;                                      // tile_dim1=1, tile_dim2=0
  g1[5] = (int)nelem;                             // tensor_dim0_stride lo32
  g1[6] = 0;
  g1[7] = 0;
  v4i z4 = {0, 0, 0, 0};
#if __clang_major__ >= 23
  v8i z8 = {0, 0, 0, 0, 0, 0, 0, 0};
  __builtin_amdgcn_tensor_load_to_lds(g0, g1, z4, z4, z8, 0);
#else
  __builtin_amdgcn_tensor_load_to_lds(g0, g1, z4, z4, 0);
#endif
}

// ---------------------------------------------------------------------------
// Weight pre-pack into WMMA A-fragment layout (f16, 32B contiguous per lane).
// ---------------------------------------------------------------------------
__global__ void k_pack_w(const float* __restrict__ src, _Float16* __restrict__ dst,
                         int taps) {
  int idx = blockIdx.x * 256 + threadIdx.x;
  int total = taps * 2 * 4 * 512;
  if (idx >= total) return;
  int h = idx & 15, lane = (idx >> 4) & 31, mtile = (idx >> 9) & 3,
      chunk = (idx >> 11) & 1, tap = idx >> 12;
  int co = mtile * 16 + (lane & 15);
  int ci = chunk * 32 + klocal(lane, h);
  dst[idx] = (_Float16)src[(co * 64 + ci) * taps + tap];
}

// adj (B,N,N) fp32 -> A fragments of adj^T (A[m,n] = adj[n,m])
__global__ void k_pack_adj(const float* __restrict__ adj, _Float16* __restrict__ dst) {
  int idx = blockIdx.x * 256 + threadIdx.x;  // total = 64*11*6*512
  int h = idx & 15, lane = (idx >> 4) & 31;
  int r = idx >> 9;
  int chunk = r % 6; r /= 6;
  int mtile = r % 11; int b = r / 11;
  int m = mtile * 16 + (lane & 15);
  int nn = chunk * 32 + klocal(lane, h);
  float v = (m < NN && nn < NN) ? adj[((long)b * NN + nn) * NN + m] : 0.f;
  dst[idx] = (_Float16)v;
}

// zero helper (uint4 granularity)
__global__ void k_zero16(uint4* __restrict__ p) {
  long i = (long)blockIdx.x * 256 + threadIdx.x;
  p[i] = make_uint4(0u, 0u, 0u, 0u);
}

// ---------------------------------------------------------------------------
// Fused tsconv (conv1x5+leaky, conv1x3+tanh) + dgcn conv1x1.  One WG per
// (b,n); 8 waves = 8 tiles.  Emits t (skip, f32), y (f32) and yT (f16 in
// the exact [c][l][n192] LDS image k_adjapply consumes via TDM).
// ---------------------------------------------------------------------------
__global__ __launch_bounds__(256)
void k_tsconv(const float* __restrict__ xin, int Ltot, int lstr, int loff,
              const _Float16* __restrict__ W1p, const float* __restrict__ b1,
              const _Float16* __restrict__ W2p, const float* __restrict__ b2,
              const _Float16* __restrict__ Cwp, const float* __restrict__ cb,
              float* __restrict__ t, float* __restrict__ y,
              _Float16* __restrict__ yT) {
  __shared__ _Float16 xlds[36 * 64];
  __shared__ _Float16 o1[36 * 64];
  __shared__ _Float16 o2[32 * 64];
  const int bid = blockIdx.x;
  const int b = bid / NN, n = bid % NN;
  const int tid = threadIdx.x, lane = tid & 31, w = tid >> 5;
  const int mt = w & 3, lt = w >> 2;
  const int col = lt * 16 + (lane & 15);
  const int cobase = mt * 16 + ((lane >> 4) << 3);

  __builtin_prefetch(W1p, 0, 3);
  for (int i = tid; i < 36 * 64; i += 256) { xlds[i] = (_Float16)0.f; o1[i] = (_Float16)0.f; }
  __syncthreads();
  for (int i = tid; i < 64 * 30; i += 256) {
    int ci = i / 30, lp = i % 30;
    int ls = lp - 3; ls = ls < 0 ? 0 : (ls > 23 ? 23 : ls);
    float v = xin[((long)(b * 64 + ci) * NN + n) * Ltot + ls * lstr + loff];
    xlds[lp * 64 + ci] = (_Float16)v;
  }
  __syncthreads();

  // ---- conv 1x5 + bias + leaky(0.01) ----
  {
    v8f acc = {};
#pragma unroll
    for (int k = 0; k < 5; ++k)
#pragma unroll
      for (int ch = 0; ch < 2; ++ch) {
        v16h a = *(const v16h*)(W1p + ((k * 2 + ch) * 4 + mt) * 512 + lane * 16);
        v16h bf = *(const v16h*)(&xlds[(col + k) * 64 + ch * 32 + ((lane >> 4) << 4)]);
        acc = wmma_f16(a, bf, acc);
      }
    v8h hv;
#pragma unroll
    for (int r = 0; r < 8; ++r) {
      float v = acc[r] + b1[cobase + r];
      v = v > 0.f ? v : 0.01f * v;
      hv[r] = (_Float16)v;
    }
    *(v8h*)(&o1[col * 64 + cobase]) = hv;
  }
  __syncthreads();

  // ---- conv 1x3 + bias + tanh -> t ----
  {
    v8f acc = {};
#pragma unroll
    for (int k = 0; k < 3; ++k)
#pragma unroll
      for (int ch = 0; ch < 2; ++ch) {
        v16h a = *(const v16h*)(W2p + ((k * 2 + ch) * 4 + mt) * 512 + lane * 16);
        v16h bf = *(const v16h*)(&o1[(col + k) * 64 + ch * 32 + ((lane >> 4) << 4)]);
        acc = wmma_f16(a, bf, acc);
      }
    v8h hv;
#pragma unroll
    for (int r = 0; r < 8; ++r) {
      float v = tanhf(acc[r] + b2[cobase + r]);
      hv[r] = (_Float16)v;
      if (col < LH) t[((long)(b * 64 + cobase + r) * NN + n) * LH + col] = v;
    }
    *(v8h*)(&o2[col * 64 + cobase]) = hv;
  }
  __syncthreads();

  // ---- dgcn conv1x1 + bias -> y (f32) and yT (f16, async-friendly) ----
  {
    v8f acc = {};
#pragma unroll
    for (int ch = 0; ch < 2; ++ch) {
      v16h a = *(const v16h*)(Cwp + (ch * 4 + mt) * 512 + lane * 16);
      v16h bf = *(const v16h*)(&o2[col * 64 + ch * 32 + ((lane >> 4) << 4)]);
      acc = wmma_f16(a, bf, acc);
    }
    if (col < LH) {
#pragma unroll
      for (int r = 0; r < 8; ++r) {
        int co = cobase + r;
        float v = acc[r] + cb[co];
        y[((long)(b * 64 + co) * NN + n) * LH + col] = v;
        yT[((long)(b * 64 + co) * LH + col) * 192 + n] = (_Float16)v;
      }
    }
  }
}

// xs[b,c,n] = sum_l y[b,c,n,l]
__global__ void k_xs(const float* __restrict__ y, float* __restrict__ xs) {
  int i = blockIdx.x * 256 + threadIdx.x;
  if (i >= BB * CC * NN) return;
  const float4* p = (const float4*)(y + (long)i * LH);
  float s = 0.f;
#pragma unroll
  for (int q = 0; q < 6; ++q) { float4 v = p[q]; s += v.x + v.y + v.z + v.w; }
  xs[i] = s;
}

// ---------------------------------------------------------------------------
// graph_gen: xs_b staged to LDS via async b128 copies, one wave per row n,
// double softmax, exact top-136 threshold via bitwise radix-select.
// ---------------------------------------------------------------------------
__global__ __launch_bounds__(256)
void k_graph(const float* __restrict__ xs, const float* __restrict__ mem,
             const float* __restrict__ fcw, const float* __restrict__ fcb,
             float* __restrict__ adj) {
  __shared__ float xl[64 * NN];  // 43,520 B, [c][m]
  const int b = blockIdx.x / 22, rc = blockIdx.x % 22;
  const int tid = threadIdx.x, lane = tid & 31, w = tid >> 5;
  {
    unsigned lbase = (unsigned)(size_t)(&xl[0]);
    const char* g = (const char*)(xs + (long)b * 64 * NN);
    for (int i = tid; i < (64 * NN * 4) / 16; i += 256)
      async_copy_b128(lbase + i * 16, g + i * 16);
    wait_asynccnt0();
  }
  __syncthreads();

  const int n = rc * 8 + w;
  const bool rv = (n < NN);
  const int nn = rv ? n : 0;
  float p1[6], p2[6], af[6];
  bool val[6];
  int ms[6];
#pragma unroll
  for (int j = 0; j < 6; ++j) {
    int m = lane + 32 * j;
    val[j] = rv && (m < NN);
    ms[j] = (m < NN) ? m : 0;
  }
#pragma unroll
  for (int j = 0; j < 6; ++j) {
    float s1 = 0.f, s2 = 0.f;
    for (int c = 0; c < 64; ++c) {
      float xc = xl[c * NN + nn];
      s1 += xc * mem[c * NN + ms[j]];
      s2 += xc * xl[c * NN + ms[j]];
    }
    p1[j] = fmaxf(s1, 0.f) * 0.125f;  // relu(s * 1/sqrt(64))
    p2[j] = fmaxf(s2, 0.f) * 0.125f;
  }
#pragma unroll
  for (int pass = 0; pass < 2; ++pass) {
    float* p = pass ? p2 : p1;
    float mx = -1e30f;
#pragma unroll
    for (int j = 0; j < 6; ++j) if (val[j]) mx = fmaxf(mx, p[j]);
    mx = wmaxf(mx);
    float s = 0.f;
#pragma unroll
    for (int j = 0; j < 6; ++j) { float e = val[j] ? __expf(p[j] - mx) : 0.f; p[j] = e; s += e; }
    s = wsumf(s);
    float inv = 1.f / s;
#pragma unroll
    for (int j = 0; j < 6; ++j) p[j] *= inv;
  }
  const float fw0 = fcw[0], fw1 = fcw[1], fb = fcb[0];
#pragma unroll
  for (int j = 0; j < 6; ++j) af[j] = p1[j] * fw0 + p2[j] * fw1 + fb;
  {
    float mx = -1e30f;
#pragma unroll
    for (int j = 0; j < 6; ++j) if (val[j]) mx = fmaxf(mx, af[j]);
    mx = wmaxf(mx);
    float s = 0.f;
#pragma unroll
    for (int j = 0; j < 6; ++j) { float e = val[j] ? __expf(af[j] - mx) : 0.f; af[j] = e; s += e; }
    s = wsumf(s);
    float inv = 1.f / s;
#pragma unroll
    for (int j = 0; j < 6; ++j) af[j] *= inv;
  }
  unsigned pref = 0u;
  for (int bit = 30; bit >= 0; --bit) {
    unsigned cand = pref | (1u << bit);
    int cnt = 0;
#pragma unroll
    for (int j = 0; j < 6; ++j)
      if (val[j] && __float_as_uint(af[j]) >= cand) cnt++;
    cnt = wsumi(cnt);
    if (cnt >= TOPK_) pref = cand;
  }
  if (rv) {
#pragma unroll
    for (int j = 0; j < 6; ++j)
      if (val[j]) {
        float o = (__float_as_uint(af[j]) >= pref) ? af[j] : 0.f;
        adj[((long)b * NN + n) * NN + (lane + 32 * j)] = o;
      }
  }
}

// ---------------------------------------------------------------------------
// z[b,c,m,l] = sum_n adj[b,n,m] * y[b,c,n,l].  One WG per (b,c).
// yT block (24x192 f16 = 9216 B, contiguous) staged via TDM tensor load.
// Output written as zT f16 in the [n][l][ci] image k_out consumes async.
// ---------------------------------------------------------------------------
__global__ __launch_bounds__(256)
void k_adjapply(const _Float16* __restrict__ yT, const _Float16* __restrict__ adjp,
                _Float16* __restrict__ zT) {
  __shared__ _Float16 yl[24 * 192];  // exact image of yT[b][c]
  const int b = blockIdx.x >> 6, c = blockIdx.x & 63;
  const int tid = threadIdx.x, lane = tid & 31, w = tid >> 5;
  if (w == 0) {
    tdm_load_1d_f16((unsigned)(size_t)(&yl[0]),
                    yT + ((long)(b * 64 + c) * LH) * 192, 24 * 192);
    __builtin_amdgcn_s_wait_tensorcnt(0);
  }
  __syncthreads();
  for (int tIdx = w; tIdx < 22; tIdx += 8) {
    int mt = tIdx >> 1, lt = tIdx & 1;
    v8f acc = {};
#pragma unroll
    for (int ch = 0; ch < 6; ++ch) {
      v16h a = *(const v16h*)(adjp + ((long)((b * 11 + mt) * 6 + ch)) * 512 + lane * 16);
      v16h bf = *(const v16h*)(&yl[(lt * 16 + (lane & 15)) * 192 + ch * 32 + ((lane >> 4) << 4)]);
      acc = wmma_f16(a, bf, acc);
    }
    int col = lt * 16 + (lane & 15);
    int mbase = mt * 16 + ((lane >> 4) << 3);
    if (col < LH) {
#pragma unroll
      for (int r = 0; r < 8; ++r) {
        int m = mbase + r;
        if (m < NN)
          zT[(((long)(b * NN + m)) * LH + col) * 64 + c] = (_Float16)acc[r];
      }
    }
  }
}

// ---------------------------------------------------------------------------
// Second conv1x1 (+bias) -> *emb + skip(t) -> branch epilogue.
// zT block (24x64 f16 = 3072 B) staged via async b128 copies.
// ---------------------------------------------------------------------------
__global__ __launch_bounds__(256)
void k_out(const _Float16* __restrict__ zT, const _Float16* __restrict__ Cwp,
           const float* __restrict__ cb, const float* __restrict__ emb,
           const float* __restrict__ skip, const float* __restrict__ aux,
           int auxLtot, int auxstr, int auxoff, int post,
           float* __restrict__ out) {
  __shared__ _Float16 zl[32 * 64];  // rows 0..23 async-filled, 24..31 zeroed
  const int bid = blockIdx.x;
  const int b = bid / NN, n = bid % NN;
  const int tid = threadIdx.x, lane = tid & 31, w = tid >> 5;
  for (int i = 24 * 64 + tid; i < 32 * 64; i += 256) zl[i] = (_Float16)0.f;
  {
    unsigned lbase = (unsigned)(size_t)(&zl[0]);
    const char* g = (const char*)(zT + ((long)(b * NN + n)) * LH * 64);
    for (int i = tid; i < (LH * 64 * 2) / 16; i += 256)  // 192 chunks
      async_copy_b128(lbase + i * 16, g + i * 16);
    wait_asynccnt0();
  }
  __syncthreads();
  const int mt = w & 3, lt = w >> 2;
  v8f acc = {};
#pragma unroll
  for (int ch = 0; ch < 2; ++ch) {
    v16h a = *(const v16h*)(Cwp + (ch * 4 + mt) * 512 + lane * 16);
    v16h bf = *(const v16h*)(&zl[(lt * 16 + (lane & 15)) * 64 + ch * 32 + ((lane >> 4) << 4)]);
    acc = wmma_f16(a, bf, acc);
  }
  const int col = lt * 16 + (lane & 15);
  if (col < LH) {
    const int cobase = mt * 16 + ((lane >> 4) << 3);
#pragma unroll
    for (int r = 0; r < 8; ++r) {
      int co = cobase + r;
      long idx = (long)(b * 64 + co) * NN + n;
      float dg = (acc[r] + cb[co]) * emb[co * NN + n] + skip[idx * LH + col];
      float a0 = aux[idx * auxLtot + col * auxstr + auxoff];
      out[idx * LH + col] = (post == 0) ? (a0 * tanhf(dg)) : (a0 + dg);
    }
  }
}

// ---------------------------------------------------------------------------
extern "C" void kernel_launch(void* const* d_in, const int* in_sizes, int n_in,
                              void* d_out, int out_size, void* d_ws, size_t ws_size,
                              hipStream_t stream) {
  const float* x     = (const float*)d_in[0];
  const float* tw1   = (const float*)d_in[1];
  const float* tb1   = (const float*)d_in[2];
  const float* tw2   = (const float*)d_in[3];
  const float* tb2   = (const float*)d_in[4];
  const float* convw = (const float*)d_in[5];
  const float* convb = (const float*)d_in[6];
  const float* mem   = (const float*)d_in[7];
  const float* fcw   = (const float*)d_in[8];
  const float* fcb   = (const float*)d_in[9];
  const float* emb   = (const float*)d_in[10];

  const long TEN  = (long)BB * CC * NN * LH;   // 16,711,680
  const long YTN  = (long)BB * CC * LH * 192;  // 18,874,368 halves
  char* ws = (char*)d_ws;
  float* t    = (float*)ws; ws += TEN * 4;
  float* y    = (float*)ws; ws += TEN * 4;
  float* cbuf = (float*)ws; ws += TEN * 4;
  float* dbuf = (float*)ws; ws += TEN * 4;
  float* xs   = (float*)ws; ws += (long)BB * CC * NN * 4;
  float* adj  = (float*)ws; ws += (long)BB * NN * NN * 4;
  _Float16* yT   = (_Float16*)ws; ws += YTN * 2;
  _Float16* zT   = (_Float16*)ws; ws += TEN * 2;
  _Float16* W1p  = (_Float16*)ws; ws += 4L * 20480 * 2;
  _Float16* W2p  = (_Float16*)ws; ws += 4L * 12288 * 2;
  _Float16* Cwp  = (_Float16*)ws; ws += 4096L * 2;
  _Float16* adjp = (_Float16*)ws; ws += 2162688L * 2;

  // zero yT once per launch (pad columns n=170..191 must be 0, stay 0)
  k_zero16<<<(int)(YTN * 2 / 16 / 256), 256, 0, stream>>>((uint4*)yT);

  for (int j = 0; j < 4; ++j) {
    k_pack_w<<<80, 256, 0, stream>>>(tw1 + (long)j * 64 * 64 * 5, W1p + (long)j * 20480, 5);
    k_pack_w<<<48, 256, 0, stream>>>(tw2 + (long)j * 64 * 64 * 3, W2p + (long)j * 12288, 3);
  }
  k_pack_w<<<16, 256, 0, stream>>>(convw, Cwp, 1);

  float* out_even = (float*)d_out;
  float* out_odd  = (float*)d_out + TEN;

  struct Branch {
    const float* in;  int Lt, str, off;
    const float* aux; int aLt, astr, aoff;
    int post; float* out;
  };
  const Branch br[4] = {
      {x,    48, 2, 0,  x,    48, 2, 1, 0, dbuf},
      {x,    48, 2, 1,  x,    48, 2, 0, 0, cbuf},
      {cbuf, 24, 1, 0,  dbuf, 24, 1, 0, 1, out_odd},
      {dbuf, 24, 1, 0,  cbuf, 24, 1, 0, 1, out_even},
  };

  for (int j = 0; j < 4; ++j) {
    const Branch& B = br[j];
    k_tsconv<<<BB * NN, 256, 0, stream>>>(B.in, B.Lt, B.str, B.off,
                                          W1p + (long)j * 20480, tb1 + j * 64,
                                          W2p + (long)j * 12288, tb2 + j * 64,
                                          Cwp, convb, t, y, yT);
    k_xs<<<(BB * CC * NN) / 256, 256, 0, stream>>>(y, xs);
    k_graph<<<BB * 22, 256, 0, stream>>>(xs, mem, fcw, fcb, adj);
    k_pack_adj<<<2162688 / 256, 256, 0, stream>>>(adj, adjp);
    k_adjapply<<<BB * CC, 256, 0, stream>>>(yT, adjp, zT);
    k_out<<<BB * NN, 256, 0, stream>>>(zT, Cwp, convb, emb, t,
                                       B.aux, B.aLt, B.astr, B.aoff, B.post, B.out);
  }
}
